// SimpleStateSpaceLayer_48507360641376
// MI455X (gfx1250) — matrix operand order repaired and checked by reading, not compiled, not run on hardware
//
#include <hip/hip_runtime.h>
#include <hip/hip_bf16.h>
#include <math.h>

// ---------------------------------------------------------------------------
// SimpleStateSpaceLayer for MI455X (gfx1250, wave32, WMMA f32_16x16x32_f16)
//
//   x_in  = u @ W_in^T + b_in                  (kernel: xin_kernel)
//   s_t   = tanh(s_{t-1} @ W_state^T + x_t)    (kernel: scan_kernel, latency)
//   y     = states @ W_out^T + u @ W_skip^T + b_out + b_skip -> LayerNorm
//                                              (kernel: out_kernel, fused)
//
// WMMA operand layout assumptions (CDNA5 ISA 7.12.2, wave32):
//   A (f16 16x32): M = lane%16, K = (h>=8?16:0) + (lane>=16?8:0) + (h%8)
//   B (f16 32x16): N = lane%16, K = (h>=8?16:0) + (lane>=16?8:0) + (h%8)
//   C (f32 16x16): N = lane%16, M = vgpr_r + (lane>=16?8:0)
// Weights are pre-swizzled so every A/B fragment load is contiguous per lane.
// ---------------------------------------------------------------------------

typedef _Float16 v16h __attribute__((ext_vector_type(16)));
typedef _Float16 v8h  __attribute__((ext_vector_type(8)));
typedef float    v8f  __attribute__((ext_vector_type(8)));
typedef float    v4f  __attribute__((ext_vector_type(4)));

__device__ __forceinline__ v8f wmma_f16(v16h a, v16h b, v8f c) {
  return __builtin_amdgcn_wmma_f32_16x16x32_f16(false, a, false, b,
                                                (short)0, c, false, false);
}

__device__ __forceinline__ v16h join16(v8h lo, v8h hi) {
  v16h r;
#pragma unroll
  for (int i = 0; i < 8; ++i) { r[i] = lo[i]; r[8 + i] = hi[i]; }
  return r;
}

// ---------------------------------------------------------------------------
// prep: pre-swizzle all weights into f16, per-lane-contiguous v16h fragments.
// blob index = (tile*32 + lane)*16 + h
// ---------------------------------------------------------------------------
__global__ void prep_kernel(const float* __restrict__ W_state,
                            const float* __restrict__ W_in,
                            const float* __restrict__ W_out,
                            const float* __restrict__ W_skip,
                            _Float16* __restrict__ wst_sw,
                            _Float16* __restrict__ win_sw,
                            _Float16* __restrict__ wout_sw,
                            _Float16* __restrict__ wskip_sw) {
  int i = blockIdx.x * 256 + threadIdx.x;
  if (i < 4096) {                       // W_state as A fragments (8 tiles)
    int tile = i >> 9, l = (i >> 4) & 31, h = i & 15;
    int mt = tile >> 1, kt = tile & 1;
    int m = mt * 16 + (l & 15);
    int k = kt * 32 + ((h >= 8) ? 16 : 0) + ((l >= 16) ? 8 : 0) + (h & 7);
    wst_sw[i] = (_Float16)W_state[m * 64 + k];
  } else if (i < 36864) {               // W_in^T as B fragments (16kc x 4nt)
    int j = i - 4096;
    int tile = j >> 9, l = (j >> 4) & 31, h = j & 15;
    int kc = tile >> 2, nt = tile & 3;
    int k = kc * 32 + ((h >= 8) ? 16 : 0) + ((l >= 16) ? 8 : 0) + (h & 7);
    int n = nt * 16 + (l & 15);
    win_sw[j] = (_Float16)W_in[n * 512 + k];
  } else if (i < 69632) {               // W_out^T as B fragments (2kc x 32nt)
    int j = i - 36864;
    int tile = j >> 9, l = (j >> 4) & 31, h = j & 15;
    int kc = tile >> 5, ntg = tile & 31;
    int k = kc * 32 + ((h >= 8) ? 16 : 0) + ((l >= 16) ? 8 : 0) + (h & 7);
    int n = ntg * 16 + (l & 15);
    wout_sw[j] = (_Float16)W_out[n * 64 + k];
  } else if (i < 331776) {              // W_skip^T as B fragments (16kc x 32nt)
    int j = i - 69632;
    int tile = j >> 9, l = (j >> 4) & 31, h = j & 15;
    int kc = tile >> 5, ntg = tile & 31;
    int k = kc * 32 + ((h >= 8) ? 16 : 0) + ((l >= 16) ? 8 : 0) + (h & 7);
    int n = ntg * 16 + (l & 15);
    wskip_sw[j] = (_Float16)W_skip[n * 512 + k];
  }
}

// ---------------------------------------------------------------------------
// x_in = u @ W_in^T + b_in, written t-major: x_ws[(t*32 + b)*64 + n]
// 512 blocks x 256 thr; each wave owns a 16-row slab x full N=64.
// ---------------------------------------------------------------------------
__global__ void xin_kernel(const float* __restrict__ u,
                           const _Float16* __restrict__ win_sw,
                           const float* __restrict__ b_in,
                           float* __restrict__ x_ws) {
  int wave = threadIdx.x >> 5;
  int lane = threadIdx.x & 31;
  int l16  = lane & 15;
  int hi   = lane >> 4;
  long R0  = (long)blockIdx.x * 128 + wave * 16;
  long row = R0 + l16;                         // A row = u row (b*2048+t)

  v8f acc[4] = {};
#pragma unroll 4
  for (int kc = 0; kc < 16; ++kc) {
    // A fragment straight from global (f32 -> f16), contiguous per lane
    const float* up = u + row * 512 + kc * 32 + hi * 8;
    __builtin_prefetch(up + 32, 0, 1);
    v4f a0 = *(const v4f*)(up);
    v4f a1 = *(const v4f*)(up + 4);
    v4f a2 = *(const v4f*)(up + 16);
    v4f a3 = *(const v4f*)(up + 20);
    v16h a;
#pragma unroll
    for (int q = 0; q < 4; ++q) {
      a[q] = (_Float16)a0[q]; a[4 + q]  = (_Float16)a1[q];
      a[8 + q] = (_Float16)a2[q]; a[12 + q] = (_Float16)a3[q];
    }
#pragma unroll
    for (int nt = 0; nt < 4; ++nt) {
      const _Float16* bp = win_sw + (((kc * 4 + nt) * 32 + lane) * 16);
      v16h b = join16(*(const v8h*)bp, *(const v8h*)(bp + 8));
      acc[nt] = wmma_f16(a, b, acc[nt]);
    }
  }
  long bidx = R0 >> 11;                        // batch
  int  t0   = (int)(R0 & 2047);                // timestep base (slab-uniform)
#pragma unroll
  for (int nt = 0; nt < 4; ++nt) {
    float bias = b_in[nt * 16 + l16];
#pragma unroll
    for (int r = 0; r < 8; ++r) {
      int t = t0 + r + hi * 8;
      x_ws[((long)t * 32 + bidx) * 64 + nt * 16 + l16] = acc[nt][r] + bias;
    }
  }
}

// ---------------------------------------------------------------------------
// Sequential scan. 1 block x 64 threads: wave nt handles batches nt*16..+15
// (fully independent). C[m,b] = W_state x s; C layout == next-step B
// orientation, transposed halves routed through a tiny padded LDS buffer.
// ---------------------------------------------------------------------------
__global__ void __launch_bounds__(64)
scan_kernel(const float* __restrict__ state0,
            const _Float16* __restrict__ wst_sw,
            const float* __restrict__ x_ws,
            float* __restrict__ st_ws,
            float* __restrict__ out_final) {
  __shared__ __align__(16) _Float16 xch[2][16 * 72];  // [wave][b-local*72 + m]
  int wave = threadIdx.x >> 5;
  int lane = threadIdx.x & 31;
  int l16  = lane & 15;
  int hi   = lane >> 4;
  int nt   = wave;                               // batch tile of this wave
  _Float16* ldsb = &xch[wave][0];

  // W_state A fragments: 4 M-tiles x 2 K-chunks, resident in registers
  v16h A[4][2];
#pragma unroll
  for (int mt = 0; mt < 4; ++mt)
#pragma unroll
    for (int kt = 0; kt < 2; ++kt) {
      const _Float16* p = wst_sw + (((mt * 2 + kt) * 32 + lane) * 16);
      A[mt][kt] = join16(*(const v8h*)p, *(const v8h*)(p + 8));
    }

  // Initial B from state0: B[k=m][n=b]
  v16h B[2];
  {
    int b = nt * 16 + l16;
#pragma unroll
    for (int kt = 0; kt < 2; ++kt) {
      v16h t;
#pragma unroll
      for (int h = 0; h < 16; ++h)
        t[h] = (_Float16)state0[b * 64 + kt * 32 + ((h >= 8) ? 16 : 0) + hi * 8 + (h & 7)];
      B[kt] = t;
    }
  }

  for (int t = 0; t < 2048; ++t) {
    const float* xp = x_ws + ((long)t * 32 + nt * 16 + l16) * 64;
    float*       sp = st_ws + ((long)t * 32 + nt * 16 + l16) * 64;
    v8f C[4];
#pragma unroll
    for (int mt = 0; mt < 4; ++mt) {            // accumulator init = x_t
      const float* q = xp + mt * 16 + hi * 8;
      v4f c0 = *(const v4f*)q;
      v4f c1 = *(const v4f*)(q + 4);
      v8f c;
#pragma unroll
      for (int i = 0; i < 4; ++i) { c[i] = c0[i]; c[4 + i] = c1[i]; }
      C[mt] = c;
    }
#pragma unroll
    for (int mt = 0; mt < 4; ++mt) {
      C[mt] = wmma_f16(A[mt][0], B[0], C[mt]);
      C[mt] = wmma_f16(A[mt][1], B[1], C[mt]);
    }
#pragma unroll
    for (int mt = 0; mt < 4; ++mt) {            // tanh, stash, transpose
      v8f c = C[mt];
      v8h hb;
#pragma unroll
      for (int r = 0; r < 8; ++r) {
        float v = tanhf(c[r]);
        c[r] = v;
        hb[r] = (_Float16)v;
      }
      *(v4f*)(sp + mt * 16 + hi * 8)     = v4f{c[0], c[1], c[2], c[3]};
      *(v4f*)(sp + mt * 16 + hi * 8 + 4) = v4f{c[4], c[5], c[6], c[7]};
      *(v8h*)(ldsb + l16 * 72 + mt * 16 + hi * 8) = hb;
      if (t == 2047) {                          // final_state -> d_out tail
        float* fp = out_final + (nt * 16 + l16) * 64 + mt * 16 + hi * 8;
        *(v4f*)fp       = v4f{c[0], c[1], c[2], c[3]};
        *(v4f*)(fp + 4) = v4f{c[4], c[5], c[6], c[7]};
      }
    }
    __syncthreads();
#pragma unroll
    for (int kt = 0; kt < 2; ++kt) {            // gather next-step B operand
      const _Float16* p = ldsb + l16 * 72 + kt * 32 + hi * 8;
      B[kt] = join16(*(const v8h*)p, *(const v8h*)(p + 16));
    }
    __syncthreads();
  }
}

// ---------------------------------------------------------------------------
// Fused output: y = u@W_skip^T (K=512) + states@W_out^T (K=64) + biases,
// LayerNorm epilogue, single write of d_out.
// 1024 blocks x 256 thr; block tile = 64 rows x 512 cols (8 waves = 2M x 4N).
// ---------------------------------------------------------------------------
#define ASTRIDE 40   // halves; 80B row stride: 16B-aligned, conflict-free
__global__ void __launch_bounds__(256)
out_kernel(const float* __restrict__ u,
           const float* __restrict__ st_ws,
           const _Float16* __restrict__ wskip_sw,
           const _Float16* __restrict__ wout_sw,
           const float* __restrict__ b_out,
           const float* __restrict__ b_skip,
           const float* __restrict__ gamma,
           const float* __restrict__ beta,
           float* __restrict__ y) {
  __shared__ __align__(16) _Float16 abuf[64 * ASTRIDE];
  __shared__ float lnsum[64][4];
  __shared__ float lnsq[64][4];
  __shared__ float lnmu[64];
  __shared__ float lnrs[64];

  int tid  = threadIdx.x;
  int wave = tid >> 5;
  int lane = tid & 31;
  int l16  = lane & 15;
  int hi   = lane >> 4;
  int wm   = wave >> 2;        // 0..1 : 32-row slab
  int wn   = wave & 3;         // 0..3 : 128-col slab
  long r0  = (long)blockIdx.x * 64;
  int  lr  = tid >> 2;         // coop-load row
  int  lc  = (tid & 3) * 8;    // coop-load col (halves)

  v8f acc[2][8] = {};

  // ---- stage 1: u @ W_skip^T, K = 512 ----
  for (int kc = 0; kc < 16; ++kc) {
    const float* up = u + (r0 + lr) * 512 + kc * 32 + lc;
    __builtin_prefetch(up + 32, 0, 1);
    v4f f0 = *(const v4f*)up;
    v4f f1 = *(const v4f*)(up + 4);
    v8h hs;
#pragma unroll
    for (int q = 0; q < 4; ++q) { hs[q] = (_Float16)f0[q]; hs[4 + q] = (_Float16)f1[q]; }
    *(v8h*)(abuf + lr * ASTRIDE + lc) = hs;
    __syncthreads();
    v16h Af[2];
#pragma unroll
    for (int mt = 0; mt < 2; ++mt) {
      const _Float16* p = abuf + (wm * 32 + mt * 16 + l16) * ASTRIDE + hi * 8;
      Af[mt] = join16(*(const v8h*)p, *(const v8h*)(p + 16));
    }
#pragma unroll
    for (int ntl = 0; ntl < 8; ++ntl) {
      int ntg = wn * 8 + ntl;
      const _Float16* bp = wskip_sw + (((long)(kc * 32 + ntg) * 32 + lane) * 16);
      v16h b = join16(*(const v8h*)bp, *(const v8h*)(bp + 8));
      acc[0][ntl] = wmma_f16(Af[0], b, acc[0][ntl]);
      acc[1][ntl] = wmma_f16(Af[1], b, acc[1][ntl]);
    }
    __syncthreads();
  }

  // ---- stage 2: states @ W_out^T, K = 64 ----
  long bidx = r0 >> 11;
  int  t0   = (int)(r0 & 2047);
  for (int kc = 0; kc < 2; ++kc) {
    const float* sp = st_ws + ((long)(t0 + lr) * 32 + bidx) * 64 + kc * 32 + lc;
    v4f f0 = *(const v4f*)sp;
    v4f f1 = *(const v4f*)(sp + 4);
    v8h hs;
#pragma unroll
    for (int q = 0; q < 4; ++q) { hs[q] = (_Float16)f0[q]; hs[4 + q] = (_Float16)f1[q]; }
    *(v8h*)(abuf + lr * ASTRIDE + lc) = hs;
    __syncthreads();
    v16h Af[2];
#pragma unroll
    for (int mt = 0; mt < 2; ++mt) {
      const _Float16* p = abuf + (wm * 32 + mt * 16 + l16) * ASTRIDE + hi * 8;
      Af[mt] = join16(*(const v8h*)p, *(const v8h*)(p + 16));
    }
#pragma unroll
    for (int ntl = 0; ntl < 8; ++ntl) {
      int ntg = wn * 8 + ntl;
      const _Float16* bp = wout_sw + (((long)(kc * 32 + ntg) * 32 + lane) * 16);
      v16h b = join16(*(const v8h*)bp, *(const v8h*)(bp + 8));
      acc[0][ntl] = wmma_f16(Af[0], b, acc[0][ntl]);
      acc[1][ntl] = wmma_f16(Af[1], b, acc[1][ntl]);
    }
    __syncthreads();
  }

  // ---- stage 3: biases + LayerNorm epilogue ----
  float biasv[8], gam[8], bet[8];
#pragma unroll
  for (int ntl = 0; ntl < 8; ++ntl) {
    int col = wn * 128 + ntl * 16 + l16;
    biasv[ntl] = b_out[col] + b_skip[col];
    gam[ntl]   = gamma[col];
    bet[ntl]   = beta[col];
  }
#pragma unroll
  for (int mt = 0; mt < 2; ++mt)
#pragma unroll
    for (int ntl = 0; ntl < 8; ++ntl)
#pragma unroll
      for (int r = 0; r < 8; ++r) acc[mt][ntl][r] += biasv[ntl];

  // per-row partial sums: reduce over 8 n-tiles, then over the 16-lane half
#pragma unroll
  for (int mt = 0; mt < 2; ++mt)
#pragma unroll
    for (int r = 0; r < 8; ++r) {
      float s = 0.f, q = 0.f;
#pragma unroll
      for (int ntl = 0; ntl < 8; ++ntl) {
        float v = acc[mt][ntl][r];
        s += v;
        q += v * v;
      }
#pragma unroll
      for (int off = 1; off < 16; off <<= 1) {   // stays within lane-half
        s += __shfl_xor(s, off, 32);
        q += __shfl_xor(q, off, 32);
      }
      if (l16 == 0) {
        int rowl = wm * 32 + mt * 16 + r + hi * 8;
        lnsum[rowl][wn] = s;
        lnsq[rowl][wn]  = q;
      }
    }
  __syncthreads();
  if (tid < 64) {
    float s  = lnsum[tid][0] + lnsum[tid][1] + lnsum[tid][2] + lnsum[tid][3];
    float q  = lnsq[tid][0] + lnsq[tid][1] + lnsq[tid][2] + lnsq[tid][3];
    float mu = s * (1.0f / 512.0f);
    float var = q * (1.0f / 512.0f) - mu * mu;
    lnmu[tid] = mu;
    lnrs[tid] = rsqrtf(var + 1e-5f);
  }
  __syncthreads();
#pragma unroll
  for (int mt = 0; mt < 2; ++mt)
#pragma unroll
    for (int r = 0; r < 8; ++r) {
      int rowl = wm * 32 + mt * 16 + r + hi * 8;
      float mu = lnmu[rowl];
      float rs = lnrs[rowl];
      long orow = (r0 + rowl) * 512;
#pragma unroll
      for (int ntl = 0; ntl < 8; ++ntl) {
        int col = wn * 128 + ntl * 16 + l16;
        y[orow + col] = (acc[mt][ntl][r] - mu) * rs * gam[ntl] + bet[ntl];
      }
    }
}

// ---------------------------------------------------------------------------
extern "C" void kernel_launch(void* const* d_in, const int* in_sizes, int n_in,
                              void* d_out, int out_size, void* d_ws, size_t ws_size,
                              hipStream_t stream) {
  (void)in_sizes; (void)n_in; (void)out_size; (void)ws_size;
  const float* u       = (const float*)d_in[0];
  const float* state0  = (const float*)d_in[1];
  const float* W_in    = (const float*)d_in[2];
  const float* b_in    = (const float*)d_in[3];
  const float* W_state = (const float*)d_in[4];
  const float* W_out   = (const float*)d_in[5];
  const float* b_out   = (const float*)d_in[6];
  const float* W_skip  = (const float*)d_in[7];
  const float* b_skip  = (const float*)d_in[8];
  const float* gamma   = (const float*)d_in[9];
  const float* beta    = (const float*)d_in[10];
  float* y = (float*)d_out;

  char* ws = (char*)d_ws;
  float*    x_ws     = (float*)(ws);                                   // 16 MB
  float*    st_ws    = (float*)(ws + (size_t)16 * 1024 * 1024);        // 16 MB
  size_t woff        = (size_t)32 * 1024 * 1024;
  _Float16* wst_sw   = (_Float16*)(ws + woff);                         // 8 KB
  _Float16* win_sw   = (_Float16*)(ws + woff + 8192);                  // 64 KB
  _Float16* wout_sw  = (_Float16*)(ws + woff + 8192 + 65536);          // 64 KB
  _Float16* wskip_sw = (_Float16*)(ws + woff + 8192 + 65536 + 65536);  // 512 KB

  prep_kernel<<<1296, 256, 0, stream>>>(W_state, W_in, W_out, W_skip,
                                        wst_sw, win_sw, wout_sw, wskip_sw);
  xin_kernel<<<512, 256, 0, stream>>>(u, win_sw, b_in, x_ws);
  scan_kernel<<<1, 64, 0, stream>>>(state0, wst_sw, x_ws, st_ws,
                                    y + (size_t)65536 * 512);
  out_kernel<<<1024, 256, 0, stream>>>(u, st_ws, wskip_sw, wout_sw,
                                       b_out, b_skip, gamma, beta, y);
}
